// HGT_58265526337914
// MI455X (gfx1250) — compile-verified, matrix operand order checked by reading, not run on hardware
//
#include <hip/hip_runtime.h>

#define HID   128
#define HEADS 4
#define DH    32

typedef __attribute__((ext_vector_type(16))) __bf16 v16bf;
typedef __attribute__((ext_vector_type(8)))  float  v8f;

union Frag { v16bf v; unsigned u[8]; };

__device__ __forceinline__ unsigned pack_bf16(float a, float b) {
  unsigned ua = __float_as_uint(a);
  unsigned ub = __float_as_uint(b);
  ua = (ua + 0x7FFFu + ((ua >> 16) & 1u)) >> 16;   // round-to-nearest-even
  ub = (ub + 0x7FFFu + ((ub >> 16) & 1u)) >> 16;
  return (ub << 16) | (ua & 0xFFFFu);
}

__device__ __forceinline__ float gelu_exact(float x) {
  return 0.5f * x * (1.0f + erff(x * 0.7071067811865475f));
}

// ---------------------------------------------------------------------------
// WMMA GEMM with LDS double-buffered A staging.
//   KC: compile-time K (0 = runtime). Real calls: KC in {128,256,512} -> the
//   K loop fully unrolls into chains of v_wmma_f32_16x16x32_bf16.
// act: 0 none, 1 leaky_relu(0.01) on output, 2 exact GELU applied to A
// skipP != null: C = sigmoid(*skipP)*y + (1-sigmoid)*xprev  (HGT skip blend)
// Block = 128 threads = 4 waves; tile = 16 rows x 64 cols (16x16 per wave).
// A tile (shared by all 4 waves) staged in LDS as packed bf16 pairs; the next
// tile's global float4 load is issued before the current WMMA (ping-pong
// buffers, one barrier per K-step; the dscnt flush in the barrier makes the
// swap safe). M % 16 == 0, K % 32 == 0 guaranteed by the problem sizes.
// ---------------------------------------------------------------------------
template <int KC>
__global__ void gemm_wmma_kernel(const float* __restrict__ A, int lda,
                                 const float* __restrict__ B, int ldb,
                                 float* __restrict__ C, int ldc,
                                 int M, int N, int K,
                                 const float* __restrict__ bias,
                                 const float* __restrict__ xprev,
                                 const float* __restrict__ skipP,
                                 int act)
{
  __shared__ unsigned ldsA[2][16][16];         // 2 x (16 rows x 16 bf16-pair u32)

  const int tid  = threadIdx.x;
  const int lane = tid & 31;
  const int wave = tid >> 5;
  const int n0 = (blockIdx.x * 4 + wave) * 16;
  const int m0 = blockIdx.y * 16;
  const bool active = (n0 < N);                // wave-uniform

  const int half = lane >> 4;
  const int mr   = lane & 15;
  const int bcol = n0 + mr;

  const int srow = tid >> 3;                   // staging row 0..15
  const int skq  = tid & 7;                    // staging float4 index 0..7
  const float* aRow = A + (size_t)(m0 + srow) * lda + skq * 4;

  const int KK = (KC > 0) ? KC : K;

  // prologue: stage first A tile
  {
    float4 a4 = *(const float4*)aRow;
    if (act == 2) {
      a4.x = gelu_exact(a4.x); a4.y = gelu_exact(a4.y);
      a4.z = gelu_exact(a4.z); a4.w = gelu_exact(a4.w);
    }
    uint2 pk; pk.x = pack_bf16(a4.x, a4.y); pk.y = pack_bf16(a4.z, a4.w);
    *(uint2*)&ldsA[0][srow][skq * 2] = pk;
  }

  v8f acc = {0.f, 0.f, 0.f, 0.f, 0.f, 0.f, 0.f, 0.f};
  int cur = 0;
  for (int kb = 0; kb < KK; kb += 32) {
    __syncthreads();                           // ldsA[cur] visible to all
    const bool more = (kb + 32 < KK);
    float4 nx;
    if (more) nx = *(const float4*)(aRow + kb + 32);   // overlap with WMMA

    if (active) {                              // wave-uniform: EXEC all-ones
      Frag af, bf;
      // A fragment: u32 word runs {half*4+0..3} and {8+half*4+0..3}
      uint4 r1 = *(const uint4*)&ldsA[cur][mr][half * 4];
      uint4 r2 = *(const uint4*)&ldsA[cur][mr][8 + half * 4];
      af.u[0] = r1.x; af.u[1] = r1.y; af.u[2] = r1.z; af.u[3] = r1.w;
      af.u[4] = r2.x; af.u[5] = r2.y; af.u[6] = r2.z; af.u[7] = r2.w;
#pragma unroll
      for (int v = 0; v < 8; ++v) {
        const int k0 = kb + ((v >> 2) << 4) + (half << 3) + ((v & 3) << 1);
        bf.u[v] = pack_bf16(B[(k0) * ldb + bcol], B[(k0 + 1) * ldb + bcol]);
      }
      acc = __builtin_amdgcn_wmma_f32_16x16x32_bf16(
          false, af.v, false, bf.v, (short)0, acc, false, false);
    }
    if (more) {
      if (act == 2) {
        nx.x = gelu_exact(nx.x); nx.y = gelu_exact(nx.y);
        nx.z = gelu_exact(nx.z); nx.w = gelu_exact(nx.w);
      }
      uint2 pk; pk.x = pack_bf16(nx.x, nx.y); pk.y = pack_bf16(nx.z, nx.w);
      *(uint2*)&ldsA[cur ^ 1][srow][skq * 2] = pk;
    }
    cur ^= 1;
  }

  if (!active) return;
  const int n = n0 + mr;
  float beta = 0.f, invb = 0.f;
  if (skipP) { beta = 1.0f / (1.0f + __expf(-skipP[0])); invb = 1.0f - beta; }
  const float bsn = bias ? bias[n] : 0.0f;
#pragma unroll
  for (int i = 0; i < 8; ++i) {
    const int m = m0 + i + (half << 3);        // C layout: VGPR i -> M=i+8*half
    float y = acc[i] + bsn;
    if (act == 1) y = (y > 0.f) ? y : 0.01f * y;
    if (skipP)  y = beta * y + invb * xprev[m * ldc + n];
    C[m * ldc + n] = y;
  }
}

// Expand per-head [H,32,32] relation matrix into block-diagonal [128,128],
// optionally pre-scaling head h by prel[h]*scale (bakes p_rel/sqrt(D) into
// k_rel so the attention kernel computes plain dot products).
__global__ void build_blockdiag_kernel(const float* __restrict__ rel,
                                       const float* __restrict__ prel,
                                       float scale,
                                       float* __restrict__ W)
{
  int t = blockIdx.x * blockDim.x + threadIdx.x;
  if (t >= HID * HID) return;
  int r = t >> 7, c = t & 127;
  int hr = r >> 5, hc = c >> 5;
  float s = scale * (prel ? prel[hr] : 1.0f);
  W[t] = (hr == hc) ? rel[(hr * DH + (r & 31)) * DH + (c & 31)] * s : 0.0f;
}

// ---------------- CSR-by-destination build (once per call) ----------------
__global__ void zero_int_kernel(int* __restrict__ p, int n) {
  int t = blockIdx.x * blockDim.x + threadIdx.x;
  if (t < n) p[t] = 0;
}

__global__ void csr_count_kernel(const int* __restrict__ ei, int E, int dstBase,
                                 int* __restrict__ cnt) {
  int e = blockIdx.x * blockDim.x + threadIdx.x;
  if (e >= E) return;
  atomicAdd(&cnt[dstBase + ei[E + e]], 1);
}

// single-block two-level exclusive scan over nNodes counts
__global__ void csr_scan_kernel(const int* __restrict__ cnt,
                                int* __restrict__ rowptr,
                                int* __restrict__ cursor, int n) {
  __shared__ int partial[256];
  const int tid = threadIdx.x;
  const int chunk = (n + 255) / 256;
  const int base = tid * chunk;
  int s = 0;
  for (int i = 0; i < chunk; ++i) {
    int idx = base + i;
    if (idx < n) s += cnt[idx];
  }
  partial[tid] = s;
  __syncthreads();
  if (tid == 0) {
    int r = 0;
    for (int i = 0; i < 256; ++i) { int t = partial[i]; partial[i] = r; r += t; }
    rowptr[n] = r;
  }
  __syncthreads();
  int run = partial[tid];
  for (int i = 0; i < chunk; ++i) {
    int idx = base + i;
    if (idx < n) { rowptr[idx] = run; cursor[idx] = run; run += cnt[idx]; }
  }
}

__global__ void csr_fill_kernel(const int* __restrict__ ei, int E, int dstBase,
                                int srcRowBase, int* __restrict__ cursor,
                                int* __restrict__ entries) {
  int e = blockIdx.x * blockDim.x + threadIdx.x;
  if (e >= E) return;
  int pos = atomicAdd(&cursor[dstBase + ei[E + e]], 1);
  entries[pos] = srcRowBase + ei[e];           // row into unified krel/vrel
}

// ---------------------------------------------------------------------------
// Fused HGT attention: one wave32 per destination node, ZERO global atomics.
//  Phase A: lane-per-edge scores (q staged in LDS) + shfl-butterfly head max
//  Phase B: lane-per-edge sum of exp (scores spilled to scb)
//  Phase C: edges serialized, lanes split 128 dims (head = lane>>3),
//           float4 v_rel gathers, single float4 store of agg per lane.
// ---------------------------------------------------------------------------
__global__ void hgt_attn_kernel(const int* __restrict__ rowptr,
                                const int* __restrict__ entries,
                                float* __restrict__ scb,
                                const float* __restrict__ krelU,
                                const float* __restrict__ vrelU,
                                const float* __restrict__ q,
                                float* __restrict__ agg, int nNodes)
{
  __shared__ float qs[8][HID];                 // per-wave q row, 4 KB/block
  const int lane = threadIdx.x & 31;
  const int wv   = threadIdx.x >> 5;
  const int d = (blockIdx.x * blockDim.x + threadIdx.x) >> 5;
  if (d >= nNodes) return;                     // wave-uniform

  const int start = rowptr[d], end = rowptr[d + 1];
  // stage q[d] into LDS (wave-local; hardware keeps DS ops in order per wave)
  *(float4*)&qs[wv][lane * 4] = *(const float4*)(q + (size_t)d * HID + lane * 4);
  const float4* qp = (const float4*)qs[wv];

  // ---- Phase A: scores + per-head running max ----
  float mxh[4] = {-__builtin_inff(), -__builtin_inff(),
                  -__builtin_inff(), -__builtin_inff()};
  for (int i = start + lane; i < end; i += 32) {
    const int row = entries[i];
    const float4* kp = (const float4*)(krelU + (size_t)row * HID);
    __builtin_prefetch(vrelU + (size_t)row * HID, 0, 1);  // warm L2 for phase C
    float acc[4] = {0.f, 0.f, 0.f, 0.f};
#pragma unroll
    for (int j = 0; j < 32; ++j) {
      float4 a = kp[j], b = qp[j];
      acc[j >> 3] += a.x * b.x + a.y * b.y + a.z * b.z + a.w * b.w;
    }
    float4 st; st.x = acc[0]; st.y = acc[1]; st.z = acc[2]; st.w = acc[3];
    *(float4*)(scb + (size_t)i * 4) = st;      // spill scores (CSR order)
#pragma unroll
    for (int h = 0; h < 4; ++h) mxh[h] = fmaxf(mxh[h], acc[h]);
  }
#pragma unroll
  for (int m = 16; m > 0; m >>= 1)
#pragma unroll
    for (int h = 0; h < 4; ++h)
      mxh[h] = fmaxf(mxh[h], __shfl_xor(mxh[h], m, 32));

  // ---- Phase B: per-head sum of exp ----
  float sm[4] = {0.f, 0.f, 0.f, 0.f};
  for (int i = start + lane; i < end; i += 32) {
    float4 s4 = *(const float4*)(scb + (size_t)i * 4);
    sm[0] += __expf(s4.x - mxh[0]);
    sm[1] += __expf(s4.y - mxh[1]);
    sm[2] += __expf(s4.z - mxh[2]);
    sm[3] += __expf(s4.w - mxh[3]);
  }
#pragma unroll
  for (int m = 16; m > 0; m >>= 1)
#pragma unroll
    for (int h = 0; h < 4; ++h)
      sm[h] += __shfl_xor(sm[h], m, 32);

  // ---- Phase C: weighted aggregation, lanes split the 128 dims ----
  const int h = lane >> 3;                     // lane covers dims 4*lane..+3
  const float mh = mxh[h];
  const float sh = sm[h] + 1e-16f;
  float4 a4; a4.x = 0.f; a4.y = 0.f; a4.z = 0.f; a4.w = 0.f;
  for (int i = start; i < end; ++i) {
    const int row = entries[i];
    const float alpha = __expf(scb[(size_t)i * 4 + h] - mh) / sh;
    float4 vv = *(const float4*)(vrelU + (size_t)row * HID + lane * 4);
    a4.x += alpha * vv.x; a4.y += alpha * vv.y;
    a4.z += alpha * vv.z; a4.w += alpha * vv.w;
  }
  *(float4*)(agg + (size_t)d * HID + lane * 4) = a4;
}

// ---------------------------------------------------------------------------
// Host orchestration.
// Input leaf order (JAX pytree flatten, sorted dict keys):
//  0:x_sample 1:x_mrna 2:x_mirna 3..6: ei_{s2m,m2s,s2i,i2s}
//  7 + 39*L per conv layer L: Wa(3) Wk(3) Wq(3) Wv(3) a_rel(4) ba(3) bk(3)
//                             bq(3) bv(3) m_rel(4) p_rel(4) skip(3)
//    (nt dicts sorted: mirna,mrna,sample ; et dicts sorted: i2s,m2s,s2i,s2m)
//  85..90: lin {mirna W,b; mrna W,b; sample W,b}   91: out.W  92: out.b
// Output: [x_mirna | x_mrna | x_sample | logits] fp32.
// ---------------------------------------------------------------------------
extern "C" void kernel_launch(void* const* d_in, const int* in_sizes, int n_in,
                              void* d_out, int out_size, void* d_ws, size_t ws_size,
                              hipStream_t stream)
{
  (void)in_sizes; (void)n_in; (void)out_size; (void)ws_size;
  const int NN[3]  = {2000, 20000, 2000};    // internal nt order: sample,mrna,mirna
  const int IND[3] = {512, 256, 128};
  const int NB[3]  = {0, 2000, 22000};       // node segment bases
  const int ntLeaf[3] = {2, 1, 0};           // pos in sorted {mirna,mrna,sample}
  const int NTOT = 24000, E = 300000, ETOT = 4 * 300000;
  // edge types (enum order s2m,m2s,s2i,i2s)
  const int etSrc[4] = {0, 1, 0, 2};
  const int etDst[4] = {1, 0, 2, 0};
  const int etEI[4]  = {3, 4, 5, 6};
  const int etRel[4] = {3, 1, 2, 0};         // pos in sorted {i2s,m2s,s2i,s2m}
  const int srcRowBase[4] = {0, 2000, 22000, 24000};  // rows in unified krel/vrel

  float* ws = (float*)d_ws;
  size_t off = 0;
  auto alloc = [&](size_t n) { float* p = ws + off; off += n; return p; };
  float* xA = alloc((size_t)NTOT * HID);
  float* xB = alloc((size_t)NTOT * HID);
  float* kb = alloc((size_t)NTOT * HID);
  float* qb = alloc((size_t)NTOT * HID);
  float* vb = alloc((size_t)NTOT * HID);
  float* krel[4]; float* vrel[4];
  for (int et = 0; et < 4; ++et) krel[et] = alloc((size_t)NN[etSrc[et]] * HID);
  for (int et = 0; et < 4; ++et) vrel[et] = alloc((size_t)NN[etSrc[et]] * HID);
  float* scb  = alloc((size_t)ETOT * HEADS);
  float* agg  = alloc((size_t)NTOT * HID);
  float* wbd  = alloc((size_t)HID * HID);
  int* cnt     = (int*)alloc(NTOT);
  int* rowptr  = (int*)alloc(NTOT + 1);
  int* cursor  = (int*)alloc(NTOT);
  int* entries = (int*)alloc(ETOT);

  float* outF = (float*)d_out;
  float* outX[3];                 // final x per internal nt, in d_out
  outX[2] = outF;                 // mirna
  outX[1] = outF + 256000;        // mrna
  outX[0] = outF + 2816000;       // sample
  float* outLogits = outF + 3072000;

  auto F = [&](int i) { return (const float*)d_in[i]; };
  auto I = [&](int i) { return (const int*)d_in[i]; };
  auto gemm = [&](const float* A, int lda, const float* B, int ldb,
                  float* C, int ldc, int M, int N, int K,
                  const float* bias, const float* xprev, const float* skipP,
                  int act) {
    dim3 g((N + 63) / 64, (M + 15) / 16);
    if (K == 128)
      gemm_wmma_kernel<128><<<g, 128, 0, stream>>>(A, lda, B, ldb, C, ldc,
                                                   M, N, K, bias, xprev, skipP, act);
    else if (K == 256)
      gemm_wmma_kernel<256><<<g, 128, 0, stream>>>(A, lda, B, ldb, C, ldc,
                                                   M, N, K, bias, xprev, skipP, act);
    else if (K == 512)
      gemm_wmma_kernel<512><<<g, 128, 0, stream>>>(A, lda, B, ldb, C, ldc,
                                                   M, N, K, bias, xprev, skipP, act);
    else
      gemm_wmma_kernel<0><<<g, 128, 0, stream>>>(A, lda, B, ldb, C, ldc,
                                                 M, N, K, bias, xprev, skipP, act);
  };

  // ---- CSR by destination (graph is static: build once, reuse both layers)
  zero_int_kernel<<<(NTOT + 255) / 256, 256, 0, stream>>>(cnt, NTOT);
  const int eB = (E + 255) / 256;
  for (int et = 0; et < 4; ++et)
    csr_count_kernel<<<eB, 256, 0, stream>>>(I(etEI[et]), E, NB[etDst[et]], cnt);
  csr_scan_kernel<<<1, 256, 0, stream>>>(cnt, rowptr, cursor, NTOT);
  for (int et = 0; et < 4; ++et)
    csr_fill_kernel<<<eB, 256, 0, stream>>>(I(etEI[et]), E, NB[etDst[et]],
                                            srcRowBase[et], cursor, entries);

  // ---- input linear + leaky_relu(0.01) ----
  for (int nt = 0; nt < 3; ++nt) {
    gemm(F(nt), IND[nt], F(85 + 2 * ntLeaf[nt]), HID,
         xA + (size_t)NB[nt] * HID, HID, NN[nt], HID, IND[nt],
         F(86 + 2 * ntLeaf[nt]), nullptr, nullptr, 1);
  }

  // ---- HGT conv layers ----
  for (int L = 0; L < 2; ++L) {
    const int lb = 7 + 39 * L;
    float* xcur = (L == 0) ? xA : xB;

    // K/Q/V projections per node type
    for (int nt = 0; nt < 3; ++nt) {
      const float* xin = xcur + (size_t)NB[nt] * HID;
      gemm(xin, HID, F(lb + 3 + ntLeaf[nt]), HID, kb + (size_t)NB[nt] * HID, HID,
           NN[nt], HID, HID, F(lb + 19 + ntLeaf[nt]), nullptr, nullptr, 0);
      gemm(xin, HID, F(lb + 6 + ntLeaf[nt]), HID, qb + (size_t)NB[nt] * HID, HID,
           NN[nt], HID, HID, F(lb + 22 + ntLeaf[nt]), nullptr, nullptr, 0);
      gemm(xin, HID, F(lb + 9 + ntLeaf[nt]), HID, vb + (size_t)NB[nt] * HID, HID,
           NN[nt], HID, HID, F(lb + 25 + ntLeaf[nt]), nullptr, nullptr, 0);
    }

    // relation transforms as block-diagonal GEMMs; p_rel/sqrt(D) baked into
    // the a_rel weights so scores are plain dot products downstream
    for (int et = 0; et < 4; ++et) {
      const int s = etSrc[et];
      build_blockdiag_kernel<<<(HID * HID + 255) / 256, 256, 0, stream>>>(
          F(lb + 12 + etRel[et]), F(lb + 32 + etRel[et]),
          0.17677669529663687f, wbd);
      gemm(kb + (size_t)NB[s] * HID, HID, wbd, HID, krel[et], HID,
           NN[s], HID, HID, nullptr, nullptr, nullptr, 0);
      build_blockdiag_kernel<<<(HID * HID + 255) / 256, 256, 0, stream>>>(
          F(lb + 28 + etRel[et]), nullptr, 1.0f, wbd);
      gemm(vb + (size_t)NB[s] * HID, HID, wbd, HID, vrel[et], HID,
           NN[s], HID, HID, nullptr, nullptr, nullptr, 0);
    }

    // fused softmax + aggregation: one wave per dst node, no atomics
    hgt_attn_kernel<<<NTOT / 8, 256, 0, stream>>>(
        rowptr, entries, scb, krel[0], vrel[0], qb, agg, NTOT);

    // node update: x_new = beta * (gelu(agg) @ Wa + ba) + (1-beta) * x
    for (int nt = 0; nt < 3; ++nt) {
      float* dst = (L == 0) ? (xB + (size_t)NB[nt] * HID) : outX[nt];
      gemm(agg + (size_t)NB[nt] * HID, HID, F(lb + 0 + ntLeaf[nt]), HID,
           dst, HID, NN[nt], HID, HID, F(lb + 16 + ntLeaf[nt]),
           xcur + (size_t)NB[nt] * HID, F(lb + 36 + ntLeaf[nt]), 2);
    }
  }

  // ---- logits = x_sample @ out.W + out.b ----
  gemm(outX[0], HID, F(91), 32, outLogits, 32, NN[0], 32, HID,
       F(92), nullptr, nullptr, 0);
}
